// OrthoHH_50818053046550
// MI455X (gfx1250) — matrix operand order repaired and checked by reading, not compiled, not run on hardware
//
#include <hip/hip_runtime.h>

// ---------------------------------------------------------------------------
// OrthoHH: out = x @ Q^T, Q = prod_i (I - 2 v_i v_i^T)
// Phase 0: normalize Householder rows (v / (||v|| + 1e-6))        -> ws.Vn
// Phase 1: per-row Householder sweep (rows of Q are independent);
//          epilogue emits bf16 hi/lo split planes of Q            -> ws.Qhi/Qlo
// Phase 2: 131072x512x512 GEMM, bf16x3 split-precision WMMA.
//          Grid ordered N-fastest so the 8 N-blocks sharing an x tile are
//          L2-co-resident (x read from HBM once: ~0.5 GB total ~ 22 us roof).
//          B tiles (pre-split bf16, no VALU work needed) are moved by the
//          Tensor Data Mover with hardware LDS padding; A tiles are
//          register-staged and hi/lo split in VALU.
// ---------------------------------------------------------------------------

typedef __attribute__((ext_vector_type(16))) __bf16        v16bf;
typedef __attribute__((ext_vector_type(8)))  __bf16        v8bf;
typedef __attribute__((ext_vector_type(8)))  float         v8f;
typedef __attribute__((ext_vector_type(4)))  unsigned int  v4u;
typedef __attribute__((ext_vector_type(8)))  int           v8i;
typedef __attribute__((ext_vector_type(4)))  int           v4i;

#define HH_EPS 1e-6f
#define D 512
#define LDT 40   // LDS row stride (elements); 80B -> conflict-free 16B frag reads

#if defined(__HIP_DEVICE_COMPILE__) && __has_builtin(__builtin_amdgcn_tensor_load_to_lds)
#define HAVE_TDM 1
#else
#define HAVE_TDM 0
#endif

__device__ __forceinline__ unsigned short f2bf_rne(float f) {
    unsigned int u = __float_as_uint(f);
    u += 0x7fffu + ((u >> 16) & 1u);          // round-to-nearest-even
    return (unsigned short)(u >> 16);
}
__device__ __forceinline__ float bf2f(unsigned short h) {
    return __uint_as_float(((unsigned int)h) << 16);
}
__device__ __forceinline__ unsigned lds_addr_u32(const void* p) {
    return (unsigned)(size_t)(const __attribute__((address_space(3))) void*)p;
}

#if HAVE_TDM
// TDM: load a 64-row x 32-col bf16 tile (row stride D elements in global) into
// LDS with 16B padding after each 64B row -> LDS row stride LDT=40 elements.
// D# per CDNA5 ISA 8.3/8.4: count=1, type=2, data_size=2B, pad 4 DW / 16 DW.
// 2D tile (tile_dim2=0) -> descriptor groups 2/3 unused, passed as zeros.
__device__ __forceinline__ void tdm_load_tile_b(unsigned lds_byte_off,
                                                const unsigned short* gsrc) {
    const unsigned long long ga = (unsigned long long)(size_t)gsrc;
    v4u g0 = { 1u,                                   // count=1 (valid descriptor)
               lds_byte_off,                         // lds_addr
               (unsigned)(ga & 0xffffffffu),         // global_addr[31:0]
               (unsigned)(((ga >> 32) & 0x01ffffffu) // global_addr[56:32]
                          | (2u << 30)) };           // type=2 ("image")
    v8i g1 = { (int)((1u << 16)                      // data_size = 2 bytes
                     | (1u << 20)                    // pad_enable
                     | (3u << 22)                    // pad_interval: 16 DWORDs
                     | (3u << 25)),                  // pad_amount:   4 DWORDs
               (int)(((unsigned)D & 0xffffu) << 16), // tensor_dim0 lo16 (=512)
               (int)(((unsigned)D & 0xffffu) << 16), // dim0 hi16=0 | dim1 lo16
               (int)(32u << 16),                     // dim1 hi16=0 | tile_dim0=32
               64,                                   // tile_dim1=64 | tile_dim2=0
               D,                                    // tensor_dim0_stride lo32
               0, 0 };                               // stride hi / dim1_stride
    const v4i z4 = {0, 0, 0, 0};
    const v8i z8 = {0, 0, 0, 0, 0, 0, 0, 0};
    // 6-arg toolchain form: (g0, g1, g2, g3, extra, cpol)
    __builtin_amdgcn_tensor_load_to_lds(g0, g1, z4, z4, z8, 0);
}
#endif

// ---------------- Phase 0: normalize reflector rows (wave per row) ----------
__global__ __launch_bounds__(256) void hh_normalize(const float* __restrict__ V,
                                                    float* __restrict__ Vn) {
    const int lane = threadIdx.x & 31;
    const int row  = blockIdx.x * 8 + (threadIdx.x >> 5);
    const float* v = V + (size_t)row * D;
    float vv[16];
    float ss = 0.f;
#pragma unroll
    for (int j = 0; j < 16; ++j) { vv[j] = v[lane + 32 * j]; ss += vv[j] * vv[j]; }
#pragma unroll
    for (int off = 16; off > 0; off >>= 1) ss += __shfl_xor(ss, off, 32);
    const float inv = 1.f / (sqrtf(ss) + HH_EPS);
    float* o = Vn + (size_t)row * D;
#pragma unroll
    for (int j = 0; j < 16; ++j) o[lane + 32 * j] = vv[j] * inv;
}

// ---------------- Phase 1: q_r <- q_r - 2 (q_r . v_i) v_i, i = 0..511 -------
// One wave owns one row of Q (16 VGPRs); reflector staged in LDS per step.
// Epilogue stores the bf16 hi/lo split of Q (all the GEMM ever needs).
__global__ __launch_bounds__(256) void hh_apply_rows(const float* __restrict__ Vn,
                                                     unsigned short* __restrict__ Qhi,
                                                     unsigned short* __restrict__ Qlo) {
    __shared__ float vsh[D];
    const int t    = threadIdx.x;
    const int lane = t & 31;
    const int row  = blockIdx.x * 8 + (t >> 5);

    float q[16];
#pragma unroll
    for (int j = 0; j < 16; ++j) q[j] = (lane + 32 * j == row) ? 1.f : 0.f;

    for (int i = 0; i < D; ++i) {
        __syncthreads();                       // previous step's vsh reads done
        vsh[t]       = Vn[(size_t)i * D + t];
        vsh[t + 256] = Vn[(size_t)i * D + t + 256];
        __syncthreads();
        float vv[16], dot = 0.f;
#pragma unroll
        for (int j = 0; j < 16; ++j) { vv[j] = vsh[lane + 32 * j]; dot += q[j] * vv[j]; }
#pragma unroll
        for (int off = 16; off > 0; off >>= 1) dot += __shfl_xor(dot, off, 32);
        const float s = -2.f * dot;
#pragma unroll
        for (int j = 0; j < 16; ++j) q[j] += s * vv[j];
    }
#pragma unroll
    for (int j = 0; j < 16; ++j) {
        const unsigned short h = f2bf_rne(q[j]);
        const unsigned short l = f2bf_rne(q[j] - bf2f(h));
        Qhi[(size_t)row * D + lane + 32 * j] = h;
        Qlo[(size_t)row * D + lane + 32 * j] = l;
    }
}

// ---------------- Phase 2: out = x @ Q^T, bf16x3 split WMMA -----------------
// Block: 256 threads = 8 waves (4 M x 2 N). Block tile 128x64, wave tile 32x32.
// blockIdx.x = N block (fast-varying -> x tile L2 reuse), blockIdx.y = M block.
__global__ __launch_bounds__(256) void hh_gemm(const float* __restrict__ X,
                                               const unsigned short* __restrict__ Qhi,
                                               const unsigned short* __restrict__ Qlo,
                                               float* __restrict__ Out) {
    __shared__ unsigned short sAhi[128 * LDT];
    __shared__ unsigned short sAlo[128 * LDT];
    __shared__ unsigned short sBhi[64 * LDT];
    __shared__ unsigned short sBlo[64 * LDT];

    const int t     = threadIdx.x;
    const int lane  = t & 31;
    const int wv    = t >> 5;
    const int waveM = wv & 3;           // 0..3 -> M offset waveM*32
    const int waveN = wv >> 2;          // 0..1 -> N offset waveN*32
    const int half  = lane >> 4;        // lane half selects K sub-range
    const int l15   = lane & 15;        // row (A) / col (B,D) within 16
    const int j0    = blockIdx.x * 64;
    const int m0    = blockIdx.y * 128;

    // A stager indices: 32 rows x 8 float4-cols per pass
    const int arow = t >> 3;            // 0..31
    const int acol = (t & 7) * 4;       // 0,4,...,28

    const v8f vzero = {0.f, 0.f, 0.f, 0.f, 0.f, 0.f, 0.f, 0.f};
    v8f acc[2][2];
    acc[0][0] = vzero; acc[0][1] = vzero; acc[1][0] = vzero; acc[1][1] = vzero;

    auto splitStore = [&](unsigned short* shi, unsigned short* slo, int base, float4 v) {
        const unsigned short h0 = f2bf_rne(v.x), h1 = f2bf_rne(v.y);
        const unsigned short h2 = f2bf_rne(v.z), h3 = f2bf_rne(v.w);
        const unsigned short l0 = f2bf_rne(v.x - bf2f(h0));
        const unsigned short l1 = f2bf_rne(v.y - bf2f(h1));
        const unsigned short l2 = f2bf_rne(v.z - bf2f(h2));
        const unsigned short l3 = f2bf_rne(v.w - bf2f(h3));
        *reinterpret_cast<uint2*>(&shi[base]) =
            make_uint2((unsigned)h0 | ((unsigned)h1 << 16), (unsigned)h2 | ((unsigned)h3 << 16));
        *reinterpret_cast<uint2*>(&slo[base]) =
            make_uint2((unsigned)l0 | ((unsigned)l1 << 16), (unsigned)l2 | ((unsigned)l3 << 16));
    };
    auto loadFrag = [&](const unsigned short* s, int rowbase, int ka, int kb) -> v16bf {
        const v8bf a = *reinterpret_cast<const v8bf*>(&s[rowbase + ka]);
        const v8bf b = *reinterpret_cast<const v8bf*>(&s[rowbase + kb]);
        return __builtin_shufflevector(a, b, 0, 1, 2, 3, 4, 5, 6, 7,
                                             8, 9, 10, 11, 12, 13, 14, 15);
    };

    for (int kt = 0; kt < D / 32; ++kt) {
        const int k0 = kt * 32;

        // A tile: global loads into registers first
        float4 ax[4];
#pragma unroll
        for (int p = 0; p < 4; ++p)
            ax[p] = *reinterpret_cast<const float4*>(
                &X[(size_t)(m0 + arow + p * 32) * D + k0 + acol]);

        __syncthreads();                  // prior iteration's fragment reads done

#if HAVE_TDM
        // B tiles: one TDM descriptor per bf16 plane, issued by wave 0.
        if (wv == 0) {
            tdm_load_tile_b(lds_addr_u32(&sBhi[0]), &Qhi[(size_t)j0 * D + k0]);
            tdm_load_tile_b(lds_addr_u32(&sBlo[0]), &Qlo[(size_t)j0 * D + k0]);
        }
#else
        // Fallback: plain bf16 tile copy, 16B per thread per plane.
        {
            const int brow = t >> 2;            // 0..63
            const int bc8  = (t & 3) * 8;       // 0,8,16,24
            *reinterpret_cast<uint4*>(&sBhi[brow * LDT + bc8]) =
                *reinterpret_cast<const uint4*>(&Qhi[(size_t)(j0 + brow) * D + k0 + bc8]);
            *reinterpret_cast<uint4*>(&sBlo[brow * LDT + bc8]) =
                *reinterpret_cast<const uint4*>(&Qlo[(size_t)(j0 + brow) * D + k0 + bc8]);
        }
#endif

        // A tile: split to bf16 hi/lo and stage to LDS (overlaps TDM)
#pragma unroll
        for (int p = 0; p < 4; ++p)
            splitStore(sAhi, sAlo, (arow + p * 32) * LDT + acol, ax[p]);

#if HAVE_TDM
        if (wv == 0) {
#if __has_builtin(__builtin_amdgcn_s_wait_tensorcnt)
            __builtin_amdgcn_s_wait_tensorcnt(0);
#else
            asm volatile("s_wait_tensorcnt 0x0" ::: "memory");
#endif
        }
#endif
        __syncthreads();                  // tiles visible

        // fragments (A and B: K runs at half*8 and 16+half*8)
        v16bf aH[2], aL[2], bH[2], bL[2];
#pragma unroll
        for (int mi = 0; mi < 2; ++mi) {
            const int rb = (waveM * 32 + mi * 16 + l15) * LDT;
            aH[mi] = loadFrag(sAhi, rb, half * 8, 16 + half * 8);
            aL[mi] = loadFrag(sAlo, rb, half * 8, 16 + half * 8);
        }
#pragma unroll
        for (int ni = 0; ni < 2; ++ni) {
            const int rb = (waveN * 32 + ni * 16 + l15) * LDT;
            bH[ni] = loadFrag(sBhi, rb, half * 8, 16 + half * 8);
            bL[ni] = loadFrag(sBlo, rb, half * 8, 16 + half * 8);
        }

        // bf16x3: hi*hi + hi*lo + lo*hi, f32 accumulation
#pragma unroll
        for (int mi = 0; mi < 2; ++mi) {
#pragma unroll
            for (int ni = 0; ni < 2; ++ni) {
                v8f c = acc[mi][ni];
                c = __builtin_amdgcn_wmma_f32_16x16x32_bf16(
                        false, aH[mi], false, bH[ni], (short)0, c, false, false);
                c = __builtin_amdgcn_wmma_f32_16x16x32_bf16(
                        false, aH[mi], false, bL[ni], (short)0, c, false, false);
                c = __builtin_amdgcn_wmma_f32_16x16x32_bf16(
                        false, aL[mi], false, bH[ni], (short)0, c, false, false);
                acc[mi][ni] = c;
            }
        }
    }

    // epilogue: D-tile layout -> VGPR r holds rows {r, r+8} (lane halves)
#pragma unroll
    for (int mi = 0; mi < 2; ++mi) {
#pragma unroll
        for (int ni = 0; ni < 2; ++ni) {
            const int col = j0 + waveN * 32 + ni * 16 + l15;
#pragma unroll
            for (int r = 0; r < 8; ++r) {
                const int rowg = m0 + waveM * 32 + mi * 16 + half * 8 + r;
                Out[(size_t)rowg * D + col] = acc[mi][ni][r];
            }
        }
    }
}

// ---------------------------------------------------------------------------
extern "C" void kernel_launch(void* const* d_in, const int* in_sizes, int n_in,
                              void* d_out, int out_size, void* d_ws, size_t ws_size,
                              hipStream_t stream) {
    const float* x = (const float*)d_in[0];   // [131072, 512]
    const float* V = (const float*)d_in[1];   // [1, 512, 512]
    float* out = (float*)d_out;               // [131072, 512]

    float* Vn           = (float*)d_ws;                        // 1 MB
    unsigned short* Qhi = (unsigned short*)(Vn + (size_t)D*D); // 0.5 MB
    unsigned short* Qlo = Qhi + (size_t)D * D;                 // 0.5 MB

    hh_normalize <<<D / 8, 256, 0, stream>>>(V, Vn);
    hh_apply_rows<<<D / 8, 256, 0, stream>>>(Vn, Qhi, Qlo);

    dim3 grid(D / 64, 131072 / 128);          // N fast-varying -> x L2 reuse
    hh_gemm      <<<grid, 256, 0, stream>>>(x, Qhi, Qlo, out);
}
